// LiquidNeuron_29652454211816
// MI455X (gfx1250) — compile-verified, matrix operand order checked by reading, not run on hardware
//
#include <hip/hip_runtime.h>
#include <math.h>

// LiquidNeuron for MI455X (gfx1250, wave32, WMMA bf16 16x16x32).
// Two-phase when workspace allows:
//   Phase 1: U = x @ W_in  (grid-wide bf16 WMMA GEMM, U bf16 in d_ws)
//   Phase 2: persistent-RNN scan, 16 WMMA/step split into two independent
//            8-deep accumulator chains, W_h register-resident, double-buffered
//            LDS h tile (1 barrier/step), pipelined U loads.
// Fallback (ws too small): fused kernel, 24 WMMA/step in two 12-deep chains.

namespace {
constexpr int NB = 128;
constexpr int NT = 512;
constexpr int NI = 256;
constexpr int NH = 512;
// LDS strides padded so (stride in dwords) % 64 == 4 -> the 16 lanes of an
// A-fragment b128 read hit 16 distinct 4-bank groups (conflict-free).
constexpr int HS_STRIDE = 520;  // bf16 (260 dwords)
constexpr int XS_STRIDE = 264;  // bf16 (132 dwords)
}

typedef __attribute__((ext_vector_type(16))) __bf16 v16bf;
typedef __attribute__((ext_vector_type(8)))  __bf16 v8bf;
typedef __attribute__((ext_vector_type(4)))  __bf16 v4bf;
typedef __attribute__((ext_vector_type(8)))  float  v8f;
typedef __attribute__((ext_vector_type(4)))  float  v4f;

#if __has_builtin(__builtin_amdgcn_tanhf)
#define FAST_TANH(x) __builtin_amdgcn_tanhf(x)
#elif __has_builtin(__builtin_amdgcn_tanh_f32)
#define FAST_TANH(x) __builtin_amdgcn_tanh_f32(x)
#else
#define FAST_TANH(x) tanhf(x)
#endif

#define WMMA_BF16(acc, a, b)                                                   \
  acc = __builtin_amdgcn_wmma_f32_16x16x32_bf16(false, (a), false, (b),        \
                                                (short)0, (acc), false, false)

// Build a 16x32 bf16 A-fragment from a row-major bf16 LDS row.
// ISA layout: lanes 0-15 elem j -> K=kb+j, elem j+8 -> K=kb+16+j.
__device__ __forceinline__ v16bf mk_a(const __bf16* p) {
  v8bf lo = *(const v8bf*)p;
  v8bf hi = *(const v8bf*)(p + 16);
  v16bf a;
#pragma unroll
  for (int j = 0; j < 8; ++j) { a[j] = lo[j]; a[j + 8] = hi[j]; }
  return a;
}

// One-time fp32 -> bf16 load of a 32xKCH*32 B-matrix slab (lane = column n).
template <int KCH>
__device__ __forceinline__ void load_bfrags(const float* __restrict__ W, int n,
                                            int half, v16bf* frag) {
#pragma unroll
  for (int c = 0; c < KCH; ++c) {
    const int kb = c * 32 + half * 8;
#pragma unroll
    for (int j = 0; j < 8; ++j) {
      frag[c][j]     = (__bf16)W[(kb + j)      * NH + n];
      frag[c][j + 8] = (__bf16)W[(kb + 16 + j) * NH + n];
    }
  }
}

// ---------------------------------------------------------------------------
// Phase 1: U[bt, h] = x[bt, :] @ Wi   (bt = b*NT + t), U stored bf16.
// 256 blocks x 256 threads. Block = (colgroup of 256 cols) x (32 M-tiles).
// Each wave owns 2 N-tiles (32 cols): Wi B-slice register-resident (128 VGPRs).
// ---------------------------------------------------------------------------
__launch_bounds__(256)
__global__ void input_proj_gemm(const float* __restrict__ x,
                                const float* __restrict__ Wi,
                                __bf16* __restrict__ U) {
  __shared__ __align__(16) __bf16 xsb[16 * XS_STRIDE];

  const int tid  = (int)threadIdx.x;
  const int lane = tid & 31, wave = tid >> 5;
  const int half = lane >> 4, nl = lane & 15;
  const int cg = (int)blockIdx.x >> 7;   // 0..1
  const int mb = (int)blockIdx.x & 127;  // 0..127
  const int na = cg * 256 + wave * 32 + nl;
  const int nb = na + 16;

  v16bf b0f[8], b1f[8];
  load_bfrags<8>(Wi, na, half, b0f);
  load_bfrags<8>(Wi, nb, half, b1f);

  const int srow = tid >> 4;           // staging row 0..15
  const int scol = (tid & 15) * 16;    // 16 floats per thread

  for (int mt = 0; mt < 32; ++mt) {
    const long row0 = (long)(mb * 32 + mt) * 16;

    // stage x tile [16,256] -> bf16 LDS (convert once, here)
    const float* xr = x + (row0 + srow) * NI + scol;
#pragma unroll
    for (int q = 0; q < 2; ++q) {
      v4f f0 = *(const v4f*)(xr + q * 8);
      v4f f1 = *(const v4f*)(xr + q * 8 + 4);
      v8bf bb;
#pragma unroll
      for (int j = 0; j < 4; ++j) { bb[j] = (__bf16)f0[j]; bb[j + 4] = (__bf16)f1[j]; }
      *(v8bf*)&xsb[srow * XS_STRIDE + scol + q * 8] = bb;
    }
    __syncthreads();

    v8f acc0 = {}, acc1 = {};
#pragma unroll
    for (int c = 0; c < 8; ++c) {
      v16bf a = mk_a(&xsb[nl * XS_STRIDE + c * 32 + half * 8]);
      WMMA_BF16(acc0, a, b0f[c]);
      WMMA_BF16(acc1, a, b1f[c]);
    }
#pragma unroll
    for (int v = 0; v < 8; ++v) {
      const long m = row0 + v + 8 * half;   // C layout: M = v (+8 for hi half)
      U[m * NH + na] = (__bf16)acc0[v];
      U[m * NH + nb] = (__bf16)acc1[v];
    }
    __syncthreads();
  }
}

// ---------------------------------------------------------------------------
// Phase 2: the serial scan. 8 WGs x 1024 threads (32 waves); wave owns 16
// hidden cols, W_h slice in 128 VGPRs for all 512 steps. 16 WMMA/step in two
// independent 8-deep accumulator chains (XDL pipelining), single barrier/step.
// ---------------------------------------------------------------------------
__launch_bounds__(1024, 1)
__global__ void liquid_scan(const __bf16* __restrict__ U,
                            const float* __restrict__ Wh,
                            const float* __restrict__ tc,
                            float* __restrict__ out) {
  __shared__ __align__(16) __bf16 hs[2][16 * HS_STRIDE];

  const int tid  = (int)threadIdx.x;
  const int lane = tid & 31, wave = tid >> 5;
  const int half = lane >> 4, nl = lane & 15;
  const int n  = wave * 16 + nl;
  const int b0 = (int)blockIdx.x * 16;

  v16bf whf[16];
  load_bfrags<16>(Wh, n, half, whf);
  const float itau = 1.0f / tc[n];

  for (int i = tid; i < 16 * HS_STRIDE; i += 1024) hs[0][i] = (__bf16)0.0f;

  v8f hreg = {};
  float uc[8];
#pragma unroll
  for (int v = 0; v < 8; ++v)
    uc[v] = (float)U[(long)(b0 + v + 8 * half) * NT * NH + n];
  __syncthreads();

  for (int t = 0; t < NT; ++t) {
    // issue next step's u loads now; consumed after the WMMA block next iter
    float un[8];
    if (t + 1 < NT) {
#pragma unroll
      for (int v = 0; v < 8; ++v)
        un[v] = (float)U[((long)(b0 + v + 8 * half) * NT + (t + 1)) * NH + n];
    } else {
#pragma unroll
      for (int v = 0; v < 8; ++v) un[v] = 0.0f;
    }

    const __bf16* hb = hs[t & 1];
    v8f acc0 = {}, acc1 = {};   // two independent chains over even/odd chunks
#pragma unroll
    for (int c = 0; c < 8; ++c) {
      v16bf a0 = mk_a(&hb[nl * HS_STRIDE + (2 * c)     * 32 + half * 8]);
      WMMA_BF16(acc0, a0, whf[2 * c]);
      v16bf a1 = mk_a(&hb[nl * HS_STRIDE + (2 * c + 1) * 32 + half * 8]);
      WMMA_BF16(acc1, a1, whf[2 * c + 1]);
    }

    __bf16* hw = hs[(t + 1) & 1];
#pragma unroll
    for (int v = 0; v < 8; ++v) {
      float tv = FAST_TANH(acc0[v] + acc1[v] + uc[v]);
      hreg[v] = hreg[v] + (tv - hreg[v]) * itau;
      hw[(v + 8 * half) * HS_STRIDE + n] = (__bf16)hreg[v];
    }
    __syncthreads();   // single barrier: ping-pong h buffers
#pragma unroll
    for (int v = 0; v < 8; ++v) uc[v] = un[v];
  }

#pragma unroll
  for (int v = 0; v < 8; ++v)
    out[(long)(b0 + v + 8 * half) * NH + n] = hreg[v];
}

// ---------------------------------------------------------------------------
// Fallback: fused single kernel (no workspace). 24 WMMA/step in two 12-deep
// chains. bf16 LDS staging for x, double-buffered xs+hs, 1 barrier/step.
// ---------------------------------------------------------------------------
__launch_bounds__(1024, 1)
__global__ void liquid_fused(const float* __restrict__ x,
                             const float* __restrict__ Wi,
                             const float* __restrict__ Wh,
                             const float* __restrict__ tc,
                             float* __restrict__ out) {
  __shared__ __align__(16) __bf16 xsb[2][16 * XS_STRIDE];
  __shared__ __align__(16) __bf16 hs[2][16 * HS_STRIDE];

  const int tid  = (int)threadIdx.x;
  const int lane = tid & 31, wave = tid >> 5;
  const int half = lane >> 4, nl = lane & 15;
  const int n  = wave * 16 + nl;
  const int b0 = (int)blockIdx.x * 16;

  v16bf whf[16];
  load_bfrags<16>(Wh, n, half, whf);
  v16bf wif[8];
  load_bfrags<8>(Wi, n, half, wif);
  const float itau = 1.0f / tc[n];

  for (int i = tid; i < 16 * HS_STRIDE; i += 1024) hs[0][i] = (__bf16)0.0f;

  const int xrow = tid >> 6;
  const int xcol = (tid & 63) * 4;     // one float4 per thread
  const float* xg = x + (long)(b0 + xrow) * NT * NI + xcol;

  { // stage x(t=0) into buffer 0
    v4f f = *(const v4f*)xg;
    v4bf b;
#pragma unroll
    for (int j = 0; j < 4; ++j) b[j] = (__bf16)f[j];
    *(v4bf*)&xsb[0][xrow * XS_STRIDE + xcol] = b;
  }

  v8f hreg = {};
  __syncthreads();

  for (int t = 0; t < NT; ++t) {
    v4f xv;
    const bool havnext = (t + 1 < NT);
    if (havnext) xv = *(const v4f*)(xg + (long)(t + 1) * NI);  // issue early

    const __bf16* xb = xsb[t & 1];
    const __bf16* hb = hs[t & 1];
    v8f acc0 = {}, acc1 = {};
#pragma unroll
    for (int c = 0; c < 8; ++c) {   // 8 x-chunks + 8 h-chunks, interleaved
      v16bf ax = mk_a(&xb[nl * XS_STRIDE + c * 32 + half * 8]);
      WMMA_BF16(acc0, ax, wif[c]);
      v16bf ah = mk_a(&hb[nl * HS_STRIDE + c * 32 + half * 8]);
      WMMA_BF16(acc1, ah, whf[c]);
    }
#pragma unroll
    for (int c = 0; c < 4; ++c) {   // remaining 8 h-chunks, alternating chains
      v16bf a0 = mk_a(&hb[nl * HS_STRIDE + (8 + 2 * c)     * 32 + half * 8]);
      WMMA_BF16(acc0, a0, whf[8 + 2 * c]);
      v16bf a1 = mk_a(&hb[nl * HS_STRIDE + (9 + 2 * c)     * 32 + half * 8]);
      WMMA_BF16(acc1, a1, whf[9 + 2 * c]);
    }

    __bf16* hw = hs[(t + 1) & 1];
#pragma unroll
    for (int v = 0; v < 8; ++v) {
      float tv = FAST_TANH(acc0[v] + acc1[v]);
      hreg[v] = hreg[v] + (tv - hreg[v]) * itau;
      hw[(v + 8 * half) * HS_STRIDE + n] = (__bf16)hreg[v];
    }
    if (havnext) {  // stage x(t+1) into the other buffer (global load now done)
      v4bf b;
#pragma unroll
      for (int j = 0; j < 4; ++j) b[j] = (__bf16)xv[j];
      *(v4bf*)&xsb[(t + 1) & 1][xrow * XS_STRIDE + xcol] = b;
    }
    __syncthreads();
  }

#pragma unroll
  for (int v = 0; v < 8; ++v)
    out[(long)(b0 + v + 8 * half) * NH + n] = hreg[v];
}

extern "C" void kernel_launch(void* const* d_in, const int* in_sizes, int n_in,
                              void* d_out, int out_size, void* d_ws, size_t ws_size,
                              hipStream_t stream) {
  (void)in_sizes; (void)n_in; (void)out_size;
  const float* x  = (const float*)d_in[0];
  const float* wi = (const float*)d_in[1];
  const float* wh = (const float*)d_in[2];
  const float* tc = (const float*)d_in[3];
  float* out = (float*)d_out;

  const size_t u_bytes = (size_t)NB * NT * NH * sizeof(__bf16);  // 64 MB
  if (d_ws != nullptr && ws_size >= u_bytes) {
    __bf16* U = (__bf16*)d_ws;
    hipLaunchKernelGGL(input_proj_gemm, dim3(256), dim3(256), 0, stream, x, wi, U);
    hipLaunchKernelGGL(liquid_scan, dim3(NB / 16), dim3(1024), 0, stream,
                       (const __bf16*)U, wh, tc, out);
  } else {
    hipLaunchKernelGGL(liquid_fused, dim3(NB / 16), dim3(1024), 0, stream,
                       x, wi, wh, tc, out);
  }
}
